// StandardAttention_65781719106165
// MI455X (gfx1250) — compile-verified, matrix operand order checked by reading, not run on hardware
//
#include <hip/hip_runtime.h>
#include <hip/hip_bf16.h>

// ---------------------------------------------------------------------------
// MHA forward for MI455X (gfx1250, wave32, WMMA 16x16x32 f16 -> f32).
// Pipeline: f32->f16 converts -> 3x WMMA GEMM (async-LDS staged W tiles,
//           V stored transposed) -> attention kernel (WMMA scores, float4 LDS
//           softmax, probs out, packed-f16 probs in LDS, WMMA ctx split-K)
//           -> WMMA output projection (fp32 out).
// ---------------------------------------------------------------------------

typedef __attribute__((ext_vector_type(16))) _Float16 v16h;
typedef __attribute__((ext_vector_type(8)))  float    v8f;

constexpr int Bc  = 2;
constexpr int Sc  = 2048;
constexpr int Dc  = 1024;
constexpr int Hc  = 16;
constexpr int HDc = 64;
constexpr int SP    = 2052;  // f32 score row stride: 16B-aligned rows, banks 4m%64 distinct
constexpr int P16LD = 2056;  // f16 prob row stride: banks 4m%64 distinct
constexpr int WLD   = 40;    // f16 LDS W-tile row stride: banks 20n%64 distinct

union Frag16 {
    v16h h;
    unsigned int u[8];
};

// K-pair base offset inside a 16x32 (or 32x16) f16 fragment for VGPR v.
__device__ __forceinline__ int frag_k(int v) {
    return ((v & 3) << 1) + ((v >> 2) << 4);   // 0,2,4,6,16,18,20,22
}

// Load A-style fragment (16 rows x 32 K, f16, row-major source, K contiguous).
// Also used for B fragments when the source is K-contiguous per N-row.
__device__ __forceinline__ void load_frag(Frag16& f, const _Float16* __restrict__ src,
                                          int row0, int ld, int kb, int lane) {
    const int r    = row0 + (lane & 15);
    const int half = (lane >> 4) & 1;
    const _Float16* p = src + (size_t)r * ld + kb + half * 8;
#pragma unroll
    for (int v = 0; v < 8; ++v) {
        f.u[v] = *reinterpret_cast<const unsigned int*>(p + frag_k(v));
    }
}

__device__ __forceinline__ v8f wmma32(const Frag16& a, const Frag16& b, v8f c) {
    return __builtin_amdgcn_wmma_f32_16x16x32_f16(false, a.h, false, b.h,
                                                  (short)0, c, false, false);
}

// gfx1250 async copy global->LDS, 16 bytes, tracked by ASYNCcnt.
// dst VGPR = LDS byte address (low 32 bits of flat pointer), src = 64-bit VA.
__device__ __forceinline__ void async_copy_b128(const _Float16* gsrc, _Float16* ldst) {
    const unsigned loff = (unsigned)(unsigned long long)(const void*)ldst;
    const unsigned long long ga = (unsigned long long)(const void*)gsrc;
    asm volatile("global_load_async_to_lds_b128 %0, %1, off"
                 :: "v"(loff), "v"(ga) : "memory");
}

__device__ __forceinline__ void wait_async0() {
    asm volatile("s_wait_asynccnt 0x0" ::: "memory");
}

// ---------------------------------------------------------------------------
// f32 -> f16 elementwise convert
// ---------------------------------------------------------------------------
__global__ void cvt_f32_to_f16(const float* __restrict__ in, _Float16* __restrict__ out,
                               size_t n) {
    size_t i = (size_t)blockIdx.x * blockDim.x + threadIdx.x;
    size_t stride = (size_t)gridDim.x * blockDim.x;
    for (; i < n; i += stride) out[i] = (_Float16)in[i];
}

// ---------------------------------------------------------------------------
// Y = X * W^T + bias.  X: [M,K] f16, W: [N,K] f16 (row-major), bias: [N] f32.
// MODE 0: Y f32 [M,N]            (output projection -> d_out)
// MODE 1: Y f16 as [B,H,S,HD]    (Q, K)
// MODE 2: Y f16 as [B,H,HD,S]    (V transposed for ctx GEMM)
// Block: 256 threads = 8 waves; each wave a 16x64 tile (A frag reused 4x);
// block covers 128x64.  The shared 64x32 W tile is staged into LDS with
// double-buffered global_load_async_to_lds_b128 (ASYNCcnt).
// ---------------------------------------------------------------------------
template <int MODE>
__global__ __launch_bounds__(256) void gemm_xwT(const _Float16* __restrict__ X,
                                                const _Float16* __restrict__ W,
                                                const float* __restrict__ bias,
                                                void* __restrict__ outp,
                                                int M, int N, int K) {
    __shared__ _Float16 wt[2][64 * WLD];   // double-buffered W tile (padded rows)

    const int wave = threadIdx.x >> 5;
    const int lane = threadIdx.x & 31;
    const int m0 = blockIdx.y * 128 + wave * 16;
    const int n0 = blockIdx.x * 64;

    // async-copy assignment: thread t moves 16B: row = t/4 (0..63), chunk = t%4
    const int trow = threadIdx.x >> 2;
    const int tchk = (threadIdx.x & 3) * 8;          // halfs within the 32-K slice
    const _Float16* wrow = W + (size_t)(n0 + trow) * K + tchk;

    auto issue = [&](int kb, int buf) {
        async_copy_b128(wrow + kb, &wt[buf][trow * WLD + tchk]);
    };

    issue(0, 0);
    int buf = 0;

    v8f acc[4] = {};
    Frag16 a, b;
    for (int kb = 0; kb < K; kb += 32) {
        wait_async0();          // our wave's async tile writes landed
        __syncthreads();        // everyone's landed; prior reads of other buf done
        if (kb + 32 < K) issue(kb + 32, buf ^ 1);

        load_frag(a, X, m0, K, kb, lane);
        if (kb + 32 < K)
            __builtin_prefetch(X + (size_t)(m0 + (lane & 15)) * K + kb + 32, 0, 3);

        const _Float16* wt_cur = &wt[buf][0];
#pragma unroll
        for (int j = 0; j < 4; ++j) {
            load_frag(b, wt_cur, 16 * j, WLD, 0, lane);
            acc[j] = wmma32(a, b, acc[j]);
        }
        buf ^= 1;
    }

    const int nl   = lane & 15;
    const int half = (lane >> 4) & 1;
#pragma unroll
    for (int j = 0; j < 4; ++j) {
        const int n = n0 + 16 * j + nl;
        const float bs = bias[n];
#pragma unroll
        for (int r = 0; r < 8; ++r) {
            const int m = m0 + r + half * 8;
            const float val = acc[j][r] + bs;
            if (MODE == 0) {
                ((float*)outp)[(size_t)m * N + n] = val;
            } else {
                const int bb = m >> 11;           // m / S
                const int s  = m & (Sc - 1);
                const int h  = n >> 6;            // n / HD
                const int hd = n & (HDc - 1);
                _Float16* O = (_Float16*)outp;
                if (MODE == 1)
                    O[(((size_t)bb * Hc + h) * Sc + s) * HDc + hd] = (_Float16)val;
                else
                    O[(((size_t)bb * Hc + h) * HDc + hd) * Sc + s] = (_Float16)val;
            }
        }
    }
}

// ---------------------------------------------------------------------------
// Attention core: one block per (b*h, 16-row q block).
//   scores = Q K^T / 8 + mask   (WMMA, staged in LDS f32 [16][SP])
//   softmax: float4 passes, wave-shuffle reductions, 2 rows per wave
//   probs -> d_out (float4 stores) + packed f16 probs in LDS
//   ctx = probs @ V             (WMMA, split-K over waves, LDS reduce)
// ---------------------------------------------------------------------------
__global__ __launch_bounds__(256) void attn_kernel(const _Float16* __restrict__ Q,
                                                   const _Float16* __restrict__ Km,
                                                   const _Float16* __restrict__ Vt,
                                                   const float* __restrict__ mask,
                                                   float* __restrict__ probs,
                                                   _Float16* __restrict__ ctx) {
    extern __shared__ char smem_raw[];
    float*     sc   = (float*)smem_raw;                 // [16][SP] f32 scores
    _Float16*  p16  = (_Float16*)(sc + 16 * SP);        // [16][P16LD] f16 probs
    float*     ctxp = (float*)(p16 + 16 * P16LD);       // [2][16][64] split-K partials

    const int wave = threadIdx.x >> 5;
    const int lane = threadIdx.x & 31;
    const int bh = blockIdx.x;                          // 0..B*H-1
    const int b  = bh >> 4;
    const int h  = bh & 15;
    const int q0 = blockIdx.y * 16;

    const _Float16* Qh = Q  + ((size_t)bh * Sc) * HDc;
    const _Float16* Kh = Km + ((size_t)bh * Sc) * HDc;
    const _Float16* Vh = Vt + ((size_t)bh * HDc) * Sc;

    // ---- scores: each wave handles 16 of the 128 key tiles -----------------
    Frag16 aq0, aq1, bk0, bk1;
    load_frag(aq0, Qh, q0, HDc, 0,  lane);
    load_frag(aq1, Qh, q0, HDc, 32, lane);

    const int half = (lane >> 4) & 1;
    const int nl   = lane & 15;
    for (int t = wave; t < Sc / 16; t += 8) {
        const int n0 = t * 16;
        load_frag(bk0, Kh, n0, HDc, 0,  lane);
        load_frag(bk1, Kh, n0, HDc, 32, lane);
        v8f c = {};
        c = wmma32(aq0, bk0, c);
        c = wmma32(aq1, bk1, c);
        const int col = n0 + nl;
        const float mval = mask[(size_t)b * Sc + col];
#pragma unroll
        for (int r = 0; r < 8; ++r) {
            const int m = r + half * 8;
            sc[m * SP + col] = c[r] * 0.125f + mval;   // 1/sqrt(64)
        }
    }
    __syncthreads();

    // ---- softmax: wave w owns rows 2w, 2w+1; float4 passes ------------------
#pragma unroll
    for (int rr = 0; rr < 2; ++rr) {
        const int row = wave * 2 + rr;
        float* srow = sc + row * SP;

        float mx = -3.0e38f;
        for (int c = lane * 4; c < Sc; c += 128) {
            const float4 v = *reinterpret_cast<const float4*>(srow + c);
            mx = fmaxf(mx, fmaxf(fmaxf(v.x, v.y), fmaxf(v.z, v.w)));
        }
#pragma unroll
        for (int off = 16; off > 0; off >>= 1) mx = fmaxf(mx, __shfl_xor(mx, off, 32));

        float sum = 0.0f;
        for (int c = lane * 4; c < Sc; c += 128) {
            float4 v = *reinterpret_cast<const float4*>(srow + c);
            v.x = __expf(v.x - mx); v.y = __expf(v.y - mx);
            v.z = __expf(v.z - mx); v.w = __expf(v.w - mx);
            sum += (v.x + v.y) + (v.z + v.w);
            *reinterpret_cast<float4*>(srow + c) = v;
        }
#pragma unroll
        for (int off = 16; off > 0; off >>= 1) sum += __shfl_xor(sum, off, 32);
        const float inv = 1.0f / sum;

        float*     prow   = probs + ((size_t)bh * Sc + (q0 + row)) * Sc;
        _Float16*  prow16 = p16 + row * P16LD;
        for (int c = lane * 4; c < Sc; c += 128) {
            float4 v = *reinterpret_cast<const float4*>(srow + c);
            v.x *= inv; v.y *= inv; v.z *= inv; v.w *= inv;
            *reinterpret_cast<float4*>(prow + c) = v;   // global_store_b128
            union { _Float16 hh[4]; uint2 u2; } pk;
            pk.hh[0] = (_Float16)v.x; pk.hh[1] = (_Float16)v.y;
            pk.hh[2] = (_Float16)v.z; pk.hh[3] = (_Float16)v.w;
            *reinterpret_cast<uint2*>(prow16 + c) = pk.u2;  // ds_store_b64
        }
    }
    __syncthreads();

    // ---- ctx = probs16 @ V: wave w -> hd tile (w&3), K half (w>>2) ----------
    const int nt = wave & 3;
    const int kh = wave >> 2;
    const int hd0 = nt * 16;
    v8f acc = {};
    Frag16 ap, bv;
    for (int kk = 0; kk < Sc / 2; kk += 32) {
        const int kb = kh * (Sc / 2) + kk;
        load_frag(ap, p16, 0, P16LD, kb, lane);     // ds_load_b32 x8
        if (kk + 32 < Sc / 2)
            __builtin_prefetch(Vh + (size_t)(hd0 + nl) * Sc + kb + 32, 0, 3);
        load_frag(bv, Vh, hd0, Sc, kb, lane);
        acc = wmma32(ap, bv, acc);
    }
#pragma unroll
    for (int r = 0; r < 8; ++r) {
        const int m = r + half * 8;
        ctxp[(kh * 16 + m) * 64 + hd0 + nl] = acc[r];
    }
    __syncthreads();

    for (int e = threadIdx.x; e < 16 * 64; e += 256) {
        const int m  = e >> 6;
        const int hd = e & 63;
        const float val = ctxp[m * 64 + hd] + ctxp[(16 + m) * 64 + hd];
        ctx[((size_t)(b * Sc + q0 + m)) * Dc + h * HDc + hd] = (_Float16)val;
    }
}

// ---------------------------------------------------------------------------
extern "C" void kernel_launch(void* const* d_in, const int* in_sizes, int n_in,
                              void* d_out, int out_size, void* d_ws, size_t ws_size,
                              hipStream_t stream) {
    const float* hidden = (const float*)d_in[0];
    const float* mask   = (const float*)d_in[1];
    const float* Wq = (const float*)d_in[2];
    const float* bq = (const float*)d_in[3];
    const float* Wk = (const float*)d_in[4];
    const float* bk = (const float*)d_in[5];
    const float* Wv = (const float*)d_in[6];
    const float* bv = (const float*)d_in[7];
    const float* Wo = (const float*)d_in[8];
    const float* bo = (const float*)d_in[9];

    const size_t M   = (size_t)Bc * Sc;        // 4096
    const size_t DD  = (size_t)Dc * Dc;        // 1M
    const size_t MD  = M * Dc;                 // 4M

    // workspace layout (f16): X | Wq | Wk | Wv | Wo | Q | K | Vt | Ctx  (~48MB)
    _Float16* p   = (_Float16*)d_ws;
    _Float16* Xh  = p;          p += MD;
    _Float16* Wqh = p;          p += DD;
    _Float16* Wkh = p;          p += DD;
    _Float16* Wvh = p;          p += DD;
    _Float16* Woh = p;          p += DD;
    _Float16* Qh  = p;          p += MD;
    _Float16* Kh  = p;          p += MD;
    _Float16* Vth = p;          p += MD;
    _Float16* Ctx = p;          p += MD;

    float* out_f  = (float*)d_out;                         // [B,S,D]
    float* probs  = out_f + MD;                            // [B,H,S,S]

    // 1) converts
    cvt_f32_to_f16<<<1024, 256, 0, stream>>>(hidden, Xh, MD);
    cvt_f32_to_f16<<<512, 256, 0, stream>>>(Wq, Wqh, DD);
    cvt_f32_to_f16<<<512, 256, 0, stream>>>(Wk, Wkh, DD);
    cvt_f32_to_f16<<<512, 256, 0, stream>>>(Wv, Wvh, DD);
    cvt_f32_to_f16<<<512, 256, 0, stream>>>(Wo, Woh, DD);

    // 2) QKV projections (WMMA + async LDS staging)
    dim3 ggrid(Dc / 64, (unsigned)(M / 128));
    gemm_xwT<1><<<ggrid, 256, 0, stream>>>(Xh, Wqh, bq, (void*)Qh,  (int)M, Dc, Dc);
    gemm_xwT<1><<<ggrid, 256, 0, stream>>>(Xh, Wkh, bk, (void*)Kh,  (int)M, Dc, Dc);
    gemm_xwT<2><<<ggrid, 256, 0, stream>>>(Xh, Wvh, bv, (void*)Vth, (int)M, Dc, Dc);

    // 3) attention core (~200 KB dynamic LDS; CDNA5 WGP has 320 KB)
    const size_t smem = (size_t)16 * SP * sizeof(float)
                      + (size_t)16 * P16LD * sizeof(_Float16)
                      + (size_t)2 * 16 * 64 * sizeof(float);
    dim3 agrid(Bc * Hc, Sc / 16);
    attn_kernel<<<agrid, 256, smem, stream>>>(Qh, Kh, Vth, mask, probs, Ctx);

    // 4) output projection -> fp32 d_out
    gemm_xwT<0><<<ggrid, 256, 0, stream>>>(Ctx, Woh, bo, (void*)out_f, (int)M, Dc, Dc);
}
